// ActivationStateReaderV3422_84791244358058
// MI455X (gfx1250) — matrix operand order repaired and checked by reading, not compiled, not run on hardware
//
#include <hip/hip_runtime.h>
#include <math.h>

typedef __attribute__((ext_vector_type(16))) _Float16 v16h;
typedef __attribute__((ext_vector_type(8)))  float    v8f;

// ---------------- problem dims ----------------
#define BB   4
#define MM   256
#define HN   8
#define KK   16
#define SEM  768
#define TRD  1024
#define UU   64
#define HID  192
#define TOK  (BB*MM*HN)        // 8192 tokens
#define EROW (TOK*KK)          // 131072 evidence rows
#define SEMD 1542              // 2*SEM+6
#define SEMP 1568              // pad to 32
#define STD_ 776               // 4*HID+8
#define STP  800               // pad to 32

// output offsets (floats)
#define OFF_ACT 0
#define OFF_AL  (TOK*HID)              // 1572864
#define OFF_UT  (OFF_AL + TOK)
#define OFF_BE  (OFF_UT + TOK)
#define OFF_AT  (OFF_BE + TOK)        // attn [TOK,16]
#define OFF_EN  (OFF_AT + TOK*KK)
#define OFF_MX  (OFF_EN + TOK)

// ---------------- device helpers ----------------
__device__ inline float nn0(float v) { return (v == v) ? v : 0.f; }   // nan->0

__device__ inline float geluf(float x) {
    return 0.5f * x * (1.f + erff(x * 0.70710678118654752f));
}

__device__ inline float warpSum(float v) {
#pragma unroll
    for (int o = 16; o > 0; o >>= 1) v += __shfl_xor(v, o, 32);
    return v;
}

// block-wide sum for blockDim.x == 256 (8 waves); s points to shared float[8]
__device__ inline float blockSum(float v, volatile float* s) {
    int lane = threadIdx.x & 31, w = threadIdx.x >> 5;
    v = warpSum(v);
    __syncthreads();
    if (lane == 0) s[w] = v;
    __syncthreads();
    float r = 0.f;
#pragma unroll
    for (int i = 0; i < 8; ++i) r += s[i];
    return r;
}

// ---------------- weight preparation ----------------
// Fold LN affine into W1: Wf[o,d] = w1[o,d]*g[d] (f16, padded), bf[o] = b1[o] + dot(w1[o,:], b_ln)
__global__ void __launch_bounds__(256)
k_fold(const float* __restrict__ w1, const float* __restrict__ g,
       const float* __restrict__ bln, const float* __restrict__ b1,
       _Float16* __restrict__ Wf, float* __restrict__ bf, int D, int Dp) {
    __shared__ float red[8];
    int o = blockIdx.x;
    float acc = 0.f;
    for (int d = threadIdx.x; d < D; d += 256) {
        float wv = w1[(size_t)o * D + d];
        acc += wv * bln[d];
        Wf[(size_t)o * Dp + d] = (_Float16)(wv * g[d]);
    }
    for (int d = D + threadIdx.x; d < Dp; d += 256) Wf[(size_t)o * Dp + d] = (_Float16)0.f;
    acc = blockSum(acc, red);
    if (threadIdx.x == 0) bf[o] = b1[o] + acc;
}

__global__ void k_cvt16(const float* __restrict__ x, _Float16* __restrict__ y, int n) {
    int i = blockIdx.x * 256 + threadIdx.x;
    if (i < n) y[i] = (_Float16)x[i];
}

// ---------------- small preprocessing ----------------
// normalized top-k weights per token
__global__ void k_wnorm(const float* __restrict__ tw, float* __restrict__ wn) {
    int t = blockIdx.x * 256 + threadIdx.x;
    if (t >= TOK) return;
    float w[KK]; float s = 0.f;
#pragma unroll
    for (int k = 0; k < KK; ++k) { w[k] = fmaxf(tw[t * KK + k], 1e-8f); s += w[k]; }
    s = fmaxf(s, 1e-6f);
#pragma unroll
    for (int k = 0; k < KK; ++k) wn[t * KK + k] = w[k] / s;
}

// Per token: normalize 16 evidence rows, standardize (LN core) -> f16 evX,
// and weighted evidence-mean -> normalized -> emean (f32)
__global__ void __launch_bounds__(256)
k_evprep(const float* __restrict__ raw, const float* __restrict__ wn,
         _Float16* __restrict__ evX, float* __restrict__ emean) {
    __shared__ float sx[SEM];
    __shared__ float sm[SEM];
    __shared__ float red[8];
    int t = blockIdx.x;
    for (int d = threadIdx.x; d < SEM; d += 256) sm[d] = 0.f;
    __syncthreads();
    for (int k = 0; k < KK; ++k) {
        const float* x = raw + ((size_t)t * KK + k) * SEM;
        float ps = 0.f, pq = 0.f;
        for (int d = threadIdx.x; d < SEM; d += 256) {
            float v = nn0(x[d]);
            sx[d] = v; ps += v; pq += v * v;
        }
        float sum = blockSum(ps, red);
        float sq  = blockSum(pq, red);
        float norm = sqrtf(sq);
        float inv  = 1.f / fmaxf(norm, 1e-12f);
        float mu   = sum * inv / (float)SEM;
        float var  = fmaxf(sq * inv * inv / (float)SEM - mu * mu, 0.f);
        float rstd = rsqrtf(var + 1e-5f);
        float wk   = wn[t * KK + k];
        _Float16* out = evX + ((size_t)t * KK + k) * SEM;
        for (int d = threadIdx.x; d < SEM; d += 256) {
            float xn = sx[d] * inv;
            out[d] = (_Float16)((xn - mu) * rstd);
            sm[d] += wk * xn;     // each d owned by one thread -> no race
        }
        __syncthreads();
    }
    // normalize evidence mean
    float pq = 0.f;
    for (int d = threadIdx.x; d < SEM; d += 256) pq += sm[d] * sm[d];
    float sq = blockSum(pq, red);
    float inv = 1.f / fmaxf(sqrtf(sq), 1e-12f);
    for (int d = threadIdx.x; d < SEM; d += 256) emean[(size_t)t * SEM + d] = sm[d] * inv;
}

// trace hidden: LN-standardize row of 1024 -> f16
__global__ void __launch_bounds__(256)
k_trace(const float* __restrict__ x, _Float16* __restrict__ y) {
    __shared__ float sx[TRD];
    __shared__ float red[8];
    int t = blockIdx.x;
    const float* xr = x + (size_t)t * TRD;
    float ps = 0.f, pq = 0.f;
    for (int d = threadIdx.x; d < TRD; d += 256) { float v = xr[d]; sx[d] = v; ps += v; pq += v * v; }
    float sum = blockSum(ps, red), sq = blockSum(pq, red);
    float mu = sum / (float)TRD;
    float var = fmaxf(sq / (float)TRD - mu * mu, 0.f);
    float rstd = rsqrtf(var + 1e-5f);
    for (int d = threadIdx.x; d < TRD; d += 256)
        y[(size_t)t * TRD + d] = (_Float16)((sx[d] - mu) * rstd);
}

// unit context gather + standardize; also assignment_usage -> scal8 slot7
__global__ void __launch_bounds__(256)
k_unit(const float* __restrict__ pta, const float* __restrict__ um,
       const float* __restrict__ aus, _Float16* __restrict__ ucsX,
       float* __restrict__ scal8) {
    __shared__ float acc[SEM];
    __shared__ float red[8];
    int t = blockIdx.x;
    int b = t >> 11; int rem = t & 2047; int m = rem >> 3; int h = rem & 7;
    const float* ar = pta + ((size_t)b * MM + m) * UU;
    for (int d = threadIdx.x; d < SEM; d += 256) acc[d] = 0.f;
    __syncthreads();
    for (int u = 0; u < UU; ++u) {
        float a = ar[u];
        const float* ur = um + (((size_t)b * UU + u) * HN + h) * SEM;
        for (int d = threadIdx.x; d < SEM; d += 256) acc[d] += a * ur[d];
    }
    __syncthreads();
    float ps = 0.f, pq = 0.f;
    for (int d = threadIdx.x; d < SEM; d += 256) { float v = acc[d]; ps += v; pq += v * v; }
    float sum = blockSum(ps, red), sq = blockSum(pq, red);
    float mu = sum / (float)SEM;
    float var = fmaxf(sq / (float)SEM - mu * mu, 0.f);
    float rstd = rsqrtf(var + 1e-5f);
    for (int d = threadIdx.x; d < SEM; d += 256)
        ucsX[(size_t)t * SEM + d] = (_Float16)((acc[d] - mu) * rstd);
    if (threadIdx.x == 0) {
        float s = 0.f;
        for (int u = 0; u < UU; ++u) s += ar[u] * aus[((size_t)b * UU + u) * HN + h];
        scal8[t * 8 + 7] = fminf(fmaxf(s, 0.f), 1.f);
    }
}

// semantic query input build + standardize -> f16 [SEMP]; scalars -> scal8[0..6]
__global__ void __launch_bounds__(256)
k_sem(const float* __restrict__ pw, const float* __restrict__ res,
      const float* __restrict__ emean, const float* __restrict__ conf_in,
      const float* __restrict__ usage_in, const float* __restrict__ selent_in,
      const float* __restrict__ selw, _Float16* __restrict__ semX,
      float* __restrict__ scal8) {
    __shared__ float pn[SEM];
    __shared__ float dr[SEM];
    __shared__ float red[8];
    int t = blockIdx.x;
    const float* pr = pw + (size_t)t * SEM;
    const float* rr = res + (size_t)t * SEM;
    const float* er = emean + (size_t)t * SEM;
    float pq = 0.f;
    for (int d = threadIdx.x; d < SEM; d += 256) { float v = nn0(pr[d]); pn[d] = v; pq += v * v; }
    float invp = 1.f / fmaxf(sqrtf(blockSum(pq, red)), 1e-12f);
    pq = 0.f;
    for (int d = threadIdx.x; d < SEM; d += 256) { float v = nn0(rr[d]); dr[d] = v; pq += v * v; }
    float nr = sqrtf(blockSum(pq, red));
    float invr = 1.f / fmaxf(nr, 1e-12f);
    for (int d = threadIdx.x; d < SEM; d += 256) { pn[d] *= invp; dr[d] *= invr; }
    __syncthreads();
    float p1 = 0.f, p2 = 0.f, p3 = 0.f;
    for (int d = threadIdx.x; d < SEM; d += 256) {
        float e = er[d];
        p1 += pn[d] * dr[d]; p2 += e * pn[d]; p3 += e * dr[d];
    }
    float prc = fminf(fmaxf(blockSum(p1, red), -1.f), 1.f);
    float epc = fminf(fmaxf(blockSum(p2, red), -1.f), 1.f);
    float erc = fminf(fmaxf(blockSum(p3, red), -1.f), 1.f);
    float rn   = fminf(fmaxf(nr, 0.f), 1.f);
    float conf = fminf(fmaxf(conf_in[t], 0.f), 1.f);
    float usag = fminf(fmaxf(usage_in[t], 0.f), 1.f);
    float sent = fminf(fmaxf(selent_in[t], 0.f), 1.f);
    float smax = -1e30f;
    for (int k = 0; k < KK; ++k) smax = fmaxf(smax, selw[t * KK + k]);
    smax = fminf(fmaxf(smax, 0.f), 1.f);
    // LN stats over 1542-dim concat
    float ps = 0.f; pq = 0.f;
    for (int d = threadIdx.x; d < SEM; d += 256) {
        ps += pn[d] + dr[d]; pq += pn[d] * pn[d] + dr[d] * dr[d];
    }
    if (threadIdx.x == 0) {
        ps += rn + prc + epc + erc + conf + usag;
        pq += rn * rn + prc * prc + epc * epc + erc * erc + conf * conf + usag * usag;
    }
    float sum = blockSum(ps, red), sq = blockSum(pq, red);
    float mu = sum / (float)SEMD;
    float var = fmaxf(sq / (float)SEMD - mu * mu, 0.f);
    float rstd = rsqrtf(var + 1e-5f);
    _Float16* out = semX + (size_t)t * SEMP;
    for (int d = threadIdx.x; d < SEM; d += 256) {
        out[d]       = (_Float16)((pn[d] - mu) * rstd);
        out[SEM + d] = (_Float16)((dr[d] - mu) * rstd);
    }
    float sc[6] = { rn, prc, epc, erc, conf, usag };
    if (threadIdx.x < 6) out[2 * SEM + threadIdx.x] = (_Float16)((sc[threadIdx.x] - mu) * rstd);
    if (threadIdx.x >= 6 && threadIdx.x < 6 + (SEMP - SEMD)) out[SEMD + threadIdx.x - 6] = (_Float16)0.f;
    if (threadIdx.x == 0) {
        float* s8 = scal8 + t * 8;
        s8[0] = rn; s8[1] = prc; s8[2] = epc; s8[3] = erc;
        s8[4] = conf; s8[5] = sent; s8[6] = smax;
    }
}

// ---------------- WMMA GEMM: Y[R,192] = X[R,Dp](f16) @ W[192,Dp]^T + bias ----------------
// 32x32 output per wave (2x2 WMMA tiles): fragments reused -> 2 b128 loads per WMMA.
// 8 waves per block cover 256 rows; grid.y = 6 column tiles of 32.
// Fragment layouts per CDNA5 ISA 7.12.2 (A and B both interleaved-K: k = 8*half + s, 16 + 8*half + s-8).
__global__ void __launch_bounds__(256)
k_gemm(const _Float16* __restrict__ X, const _Float16* __restrict__ W,
       const float* __restrict__ bias, float* __restrict__ Yf,
       _Float16* __restrict__ Yh, int R, int Dp, int act) {
    int lane = threadIdx.x & 31;
    int wv   = threadIdx.x >> 5;
    int m0 = (blockIdx.x * 8 + wv) * 32;
    int n0 = blockIdx.y * 32;
    if (m0 >= R) return;                    // wave-uniform
    int half = lane >> 4;                   // 0 | 1
    int l15  = lane & 15;
    const _Float16* xp0 = X + (size_t)(m0 + l15) * Dp + half * 8;        // A rows tile 0
    const _Float16* xp1 = xp0 + (size_t)16 * Dp;                          // A rows tile 1
    const _Float16* wp0 = W + (size_t)(n0 + l15) * Dp + half * 8;        // B cols tile 0
    const _Float16* wp1 = wp0 + (size_t)16 * Dp;                          // B cols tile 1
    v8f c00 = { 0.f, 0.f, 0.f, 0.f, 0.f, 0.f, 0.f, 0.f };
    v8f c01 = c00, c10 = c00, c11 = c00;
    for (int kb = 0; kb < Dp; kb += 32) {
        union { uint4 u[2]; v16h v; } A0, A1, B0, B1;
        A0.u[0] = *reinterpret_cast<const uint4*>(xp0 + kb);
        A0.u[1] = *reinterpret_cast<const uint4*>(xp0 + kb + 16);
        A1.u[0] = *reinterpret_cast<const uint4*>(xp1 + kb);
        A1.u[1] = *reinterpret_cast<const uint4*>(xp1 + kb + 16);
        B0.u[0] = *reinterpret_cast<const uint4*>(wp0 + kb);
        B0.u[1] = *reinterpret_cast<const uint4*>(wp0 + kb + 16);
        B1.u[0] = *reinterpret_cast<const uint4*>(wp1 + kb);
        B1.u[1] = *reinterpret_cast<const uint4*>(wp1 + kb + 16);
        if (kb + 32 < Dp) {
            __builtin_prefetch(xp0 + kb + 32, 0, 3);
            __builtin_prefetch(xp1 + kb + 32, 0, 3);
            __builtin_prefetch(wp0 + kb + 32, 0, 3);
            __builtin_prefetch(wp1 + kb + 32, 0, 3);
        }
        c00 = __builtin_amdgcn_wmma_f32_16x16x32_f16(false, A0.v, false, B0.v, (short)0, c00, false, false);
        c01 = __builtin_amdgcn_wmma_f32_16x16x32_f16(false, A0.v, false, B1.v, (short)0, c01, false, false);
        c10 = __builtin_amdgcn_wmma_f32_16x16x32_f16(false, A1.v, false, B0.v, (short)0, c10, false, false);
        c11 = __builtin_amdgcn_wmma_f32_16x16x32_f16(false, A1.v, false, B1.v, (short)0, c11, false, false);
    }
    v8f* cc[2][2] = { { &c00, &c01 }, { &c10, &c11 } };
#pragma unroll
    for (int i = 0; i < 2; ++i) {
        int mb = m0 + i * 16 + half * 8;
#pragma unroll
        for (int j = 0; j < 2; ++j) {
            int n = n0 + j * 16 + l15;
            float bn = bias ? bias[n] : 0.f;
            v8f c = *cc[i][j];
#pragma unroll
            for (int r = 0; r < 8; ++r) {
                float v = c[r] + bn;
                if (act) v = geluf(v);
                size_t idx = (size_t)(mb + r) * HID + n;
                if (Yf) Yf[idx] = v;
                if (Yh) Yh[idx] = (_Float16)v;
            }
        }
    }
}

// ---------------- attention ----------------
__global__ void __launch_bounds__(256)
k_attn(const float* __restrict__ tq, const float* __restrict__ sq,
       const float* __restrict__ keym, const float* __restrict__ valm,
       const float* __restrict__ wn, float* __restrict__ ec,
       float* __restrict__ o_attn, float* __restrict__ o_ent, float* __restrict__ o_max) {
    __shared__ float q[HID];
    __shared__ float lg[KK];
    __shared__ float at[KK];
    int t = blockIdx.x;
    if (threadIdx.x < HID) q[threadIdx.x] = tq[(size_t)t * HID + threadIdx.x] + sq[(size_t)t * HID + threadIdx.x];
    __syncthreads();
    int lane = threadIdx.x & 31, wv = threadIdx.x >> 5;
    int half = lane >> 4, l15 = lane & 15;
    {
        int k = wv * 2 + half;
        const float* kr = keym + ((size_t)t * KK + k) * HID;
        float p = 0.f;
#pragma unroll
        for (int j = 0; j < 12; ++j) p += q[l15 * 12 + j] * kr[l15 * 12 + j];
#pragma unroll
        for (int o = 8; o > 0; o >>= 1) p += __shfl_xor(p, o, 32);
        if (l15 == 0) lg[k] = p * 0.0721687836f + logf(fmaxf(wn[t * KK + k], 1e-8f));
    }
    __syncthreads();
    if (threadIdx.x == 0) {
        float mx = -1e30f;
        for (int k = 0; k < KK; ++k) mx = fmaxf(mx, lg[k]);
        float s = 0.f;
        for (int k = 0; k < KK; ++k) { float e = expf(lg[k] - mx); at[k] = e; s += e; }
        float inv = 1.f / s;
        float ent = 0.f, ma = 0.f;
        for (int k = 0; k < KK; ++k) {
            float a = at[k] * inv; at[k] = a;
            float ac = fmaxf(a, 1e-8f);
            ent -= ac * logf(ac);
            ma = fmaxf(ma, a);
        }
        o_ent[t] = ent / 2.7725887222397811f;   // log(16)
        o_max[t] = ma;
    }
    __syncthreads();
    if (threadIdx.x < KK) o_attn[t * KK + threadIdx.x] = at[threadIdx.x];
    if (threadIdx.x < HID) {
        float s = 0.f;
#pragma unroll
        for (int k = 0; k < KK; ++k) s += at[k] * valm[((size_t)t * KK + k) * HID + threadIdx.x];
        ec[(size_t)t * HID + threadIdx.x] = s;
    }
}

// ---------------- state input build ----------------
__global__ void __launch_bounds__(256)
k_state(const float* __restrict__ tq, const float* __restrict__ sq,
        const float* __restrict__ ec, const float* __restrict__ uc,
        const float* __restrict__ scal8, _Float16* __restrict__ stX) {
    __shared__ float sv[STD_];
    __shared__ float red[8];
    int t = blockIdx.x;
    float ps = 0.f, pq = 0.f;
    for (int i = threadIdx.x; i < STD_; i += 256) {
        float v;
        if (i < 192)      v = tq[(size_t)t * HID + i];
        else if (i < 384) v = sq[(size_t)t * HID + (i - 192)];
        else if (i < 576) v = ec[(size_t)t * HID + (i - 384)];
        else if (i < 768) v = uc[(size_t)t * HID + (i - 576)];
        else              v = scal8[t * 8 + (i - 768)];
        sv[i] = v; ps += v; pq += v * v;
    }
    float sum = blockSum(ps, red), sq2 = blockSum(pq, red);
    float mu = sum / (float)STD_;
    float var = fmaxf(sq2 / (float)STD_ - mu * mu, 0.f);
    float rstd = rsqrtf(var + 1e-5f);
    _Float16* out = stX + (size_t)t * STP;
    for (int i = threadIdx.x; i < STD_; i += 256) out[i] = (_Float16)((sv[i] - mu) * rstd);
    for (int i = STD_ + threadIdx.x; i < STP; i += 256) out[i] = (_Float16)0.f;
}

// ---------------- heads ----------------
__global__ void __launch_bounds__(256)
k_heads(const float* __restrict__ act,
        const float* __restrict__ wa, const float* __restrict__ ba,
        const float* __restrict__ wu, const float* __restrict__ bu,
        const float* __restrict__ wb, const float* __restrict__ bbn,
        float* __restrict__ o_al, float* __restrict__ o_ut, float* __restrict__ o_be) {
    int lane = threadIdx.x & 31, wv = threadIdx.x >> 5;
    int t = blockIdx.x * 8 + wv;
    const float* ar = act + (size_t)t * HID;
    float s0 = 0.f, s1 = 0.f, s2 = 0.f;
#pragma unroll
    for (int j = 0; j < 6; ++j) {
        int d = lane * 6 + j;
        float a = ar[d];
        s0 += a * wa[d]; s1 += a * wu[d]; s2 += a * wb[d];
    }
    s0 = warpSum(s0); s1 = warpSum(s1); s2 = warpSum(s2);
    if (lane == 0) { o_al[t] = s0 + ba[0]; o_ut[t] = s1 + bu[0]; o_be[t] = s2 + bbn[0]; }
}

// ---------------- host launcher ----------------
extern "C" void kernel_launch(void* const* d_in, const int* in_sizes, int n_in,
                              void* d_out, int out_size, void* d_ws, size_t ws_size,
                              hipStream_t stream) {
    (void)in_sizes; (void)n_in; (void)out_size; (void)ws_size;
    // inputs
    const float* in_trace  = (const float*)d_in[0];
    const float* in_raw    = (const float*)d_in[1];
    const float* in_tw     = (const float*)d_in[2];
    const float* in_pw     = (const float*)d_in[3];
    const float* in_res    = (const float*)d_in[4];
    const float* in_um     = (const float*)d_in[5];
    const float* in_pta    = (const float*)d_in[6];
    const float* in_usage  = (const float*)d_in[7];
    const float* in_conf   = (const float*)d_in[8];
    const float* in_selent = (const float*)d_in[9];
    const float* in_selw   = (const float*)d_in[10];
    const float* in_aus    = (const float*)d_in[11];
    // params (jax pytree: dicts sorted by key)
    const float* ek_b1 = (const float*)d_in[12]; const float* ek_lnb = (const float*)d_in[13];
    const float* ek_lng = (const float*)d_in[14]; const float* ek_w1 = (const float*)d_in[15];
    const float* ev_b1 = (const float*)d_in[16]; const float* ev_b2 = (const float*)d_in[17];
    const float* ev_lnb = (const float*)d_in[18]; const float* ev_lng = (const float*)d_in[19];
    const float* ev_w1 = (const float*)d_in[20]; const float* ev_w2 = (const float*)d_in[21];
    const float* h_al_b = (const float*)d_in[22]; const float* h_al_w = (const float*)d_in[23];
    const float* h_be_b = (const float*)d_in[24]; const float* h_be_w = (const float*)d_in[25];
    const float* h_ut_b = (const float*)d_in[26]; const float* h_ut_w = (const float*)d_in[27];
    const float* sq_b1 = (const float*)d_in[28]; const float* sq_b2 = (const float*)d_in[29];
    const float* sq_lnb = (const float*)d_in[30]; const float* sq_lng = (const float*)d_in[31];
    const float* sq_w1 = (const float*)d_in[32]; const float* sq_w2 = (const float*)d_in[33];
    const float* st_b1 = (const float*)d_in[34]; const float* st_b2 = (const float*)d_in[35];
    const float* st_lnb = (const float*)d_in[36]; const float* st_lng = (const float*)d_in[37];
    const float* st_w1 = (const float*)d_in[38]; const float* st_w2 = (const float*)d_in[39];
    const float* tq_b1 = (const float*)d_in[40]; const float* tq_b2 = (const float*)d_in[41];
    const float* tq_lnb = (const float*)d_in[42]; const float* tq_lng = (const float*)d_in[43];
    const float* tq_w1 = (const float*)d_in[44]; const float* tq_w2 = (const float*)d_in[45];
    const float* uv_b1 = (const float*)d_in[46]; const float* uv_b2 = (const float*)d_in[47];
    const float* uv_lnb = (const float*)d_in[48]; const float* uv_lng = (const float*)d_in[49];
    const float* uv_w1 = (const float*)d_in[50]; const float* uv_w2 = (const float*)d_in[51];

    float* out = (float*)d_out;

    // workspace carve-up
    char* base = (char*)d_ws; size_t off = 0;
    auto alloc = [&](size_t bytes) { void* p = base + off; off += (bytes + 255) & ~(size_t)255; return p; };
    float*    wnrm  = (float*)   alloc((size_t)TOK * KK * 4);
    float*    scal8 = (float*)   alloc((size_t)TOK * 8 * 4);
    float*    emean = (float*)   alloc((size_t)TOK * SEM * 4);
    _Float16* evX   = (_Float16*)alloc((size_t)EROW * SEM * 2);
    _Float16* trX   = (_Float16*)alloc((size_t)TOK * TRD * 2);
    _Float16* smX   = (_Float16*)alloc((size_t)TOK * SEMP * 2);
    _Float16* ucX   = (_Float16*)alloc((size_t)TOK * SEM * 2);
    _Float16* stX   = (_Float16*)alloc((size_t)TOK * STP * 2);
    float*    keym  = (float*)   alloc((size_t)EROW * HID * 4);
    float*    valm  = (float*)   alloc((size_t)EROW * HID * 4);
    _Float16* evh   = (_Float16*)alloc((size_t)EROW * HID * 2);
    _Float16* th    = (_Float16*)alloc((size_t)TOK * HID * 2);
    _Float16* sh    = (_Float16*)alloc((size_t)TOK * HID * 2);
    _Float16* uh    = (_Float16*)alloc((size_t)TOK * HID * 2);
    _Float16* sth   = (_Float16*)alloc((size_t)TOK * HID * 2);
    float*    tqo   = (float*)   alloc((size_t)TOK * HID * 4);
    float*    sqo   = (float*)   alloc((size_t)TOK * HID * 4);
    float*    uco   = (float*)   alloc((size_t)TOK * HID * 4);
    float*    eco   = (float*)   alloc((size_t)TOK * HID * 4);
    _Float16* Wk    = (_Float16*)alloc((size_t)HID * SEM * 2);
    _Float16* Wv1   = (_Float16*)alloc((size_t)HID * SEM * 2);
    _Float16* Wt1   = (_Float16*)alloc((size_t)HID * TRD * 2);
    _Float16* Ws1   = (_Float16*)alloc((size_t)HID * SEMP * 2);
    _Float16* Wu1   = (_Float16*)alloc((size_t)HID * SEM * 2);
    _Float16* Wst1  = (_Float16*)alloc((size_t)HID * STP * 2);
    _Float16* Wv2   = (_Float16*)alloc((size_t)HID * HID * 2);
    _Float16* Wt2   = (_Float16*)alloc((size_t)HID * HID * 2);
    _Float16* Ws2   = (_Float16*)alloc((size_t)HID * HID * 2);
    _Float16* Wu2   = (_Float16*)alloc((size_t)HID * HID * 2);
    _Float16* Wst2  = (_Float16*)alloc((size_t)HID * HID * 2);
    float*    bk    = (float*)   alloc(HID * 4);
    float*    bv1   = (float*)   alloc(HID * 4);
    float*    bt1   = (float*)   alloc(HID * 4);
    float*    bs1   = (float*)   alloc(HID * 4);
    float*    bu1   = (float*)   alloc(HID * 4);
    float*    bst1  = (float*)   alloc(HID * 4);

    // 1) fold LN into W1 for all towers
    k_fold<<<HID, 256, 0, stream>>>(ek_w1, ek_lng, ek_lnb, ek_b1, Wk,   bk,   SEM,  SEM);
    k_fold<<<HID, 256, 0, stream>>>(ev_w1, ev_lng, ev_lnb, ev_b1, Wv1,  bv1,  SEM,  SEM);
    k_fold<<<HID, 256, 0, stream>>>(tq_w1, tq_lng, tq_lnb, tq_b1, Wt1,  bt1,  TRD,  TRD);
    k_fold<<<HID, 256, 0, stream>>>(sq_w1, sq_lng, sq_lnb, sq_b1, Ws1,  bs1,  SEMD, SEMP);
    k_fold<<<HID, 256, 0, stream>>>(uv_w1, uv_lng, uv_lnb, uv_b1, Wu1,  bu1,  SEM,  SEM);
    k_fold<<<HID, 256, 0, stream>>>(st_w1, st_lng, st_lnb, st_b1, Wst1, bst1, STD_, STP);
    int n2 = HID * HID;
    k_cvt16<<<(n2 + 255) / 256, 256, 0, stream>>>(ev_w2, Wv2, n2);
    k_cvt16<<<(n2 + 255) / 256, 256, 0, stream>>>(tq_w2, Wt2, n2);
    k_cvt16<<<(n2 + 255) / 256, 256, 0, stream>>>(sq_w2, Ws2, n2);
    k_cvt16<<<(n2 + 255) / 256, 256, 0, stream>>>(uv_w2, Wu2, n2);
    k_cvt16<<<(n2 + 255) / 256, 256, 0, stream>>>(st_w2, Wst2, n2);

    // 2) preprocessing
    k_wnorm<<<(TOK + 255) / 256, 256, 0, stream>>>(in_tw, wnrm);
    k_evprep<<<TOK, 256, 0, stream>>>(in_raw, wnrm, evX, emean);
    k_trace<<<TOK, 256, 0, stream>>>(in_trace, trX);
    k_unit<<<TOK, 256, 0, stream>>>(in_pta, in_um, in_aus, ucX, scal8);
    k_sem<<<TOK, 256, 0, stream>>>(in_pw, in_res, emean, in_conf, in_usage,
                                   in_selent, in_selw, smX, scal8);

    // 3) WMMA GEMMs  (grid: rows/256 x 6 col-tiles of 32)
    dim3 gE(EROW / 256, HID / 32), gT(TOK / 256, HID / 32);
    k_gemm<<<gE, 256, 0, stream>>>(evX, Wk,  bk,  keym, nullptr, EROW, SEM, 0);   // evidence key
    k_gemm<<<gE, 256, 0, stream>>>(evX, Wv1, bv1, nullptr, evh,  EROW, SEM, 1);   // evidence value L1+gelu
    k_gemm<<<gE, 256, 0, stream>>>(evh, Wv2, ev_b2, valm, nullptr, EROW, HID, 0); // evidence value L2
    k_gemm<<<gT, 256, 0, stream>>>(trX, Wt1, bt1, nullptr, th, TOK, TRD, 1);
    k_gemm<<<gT, 256, 0, stream>>>(th,  Wt2, tq_b2, tqo, nullptr, TOK, HID, 0);   // trace_q
    k_gemm<<<gT, 256, 0, stream>>>(smX, Ws1, bs1, nullptr, sh, TOK, SEMP, 1);
    k_gemm<<<gT, 256, 0, stream>>>(sh,  Ws2, sq_b2, sqo, nullptr, TOK, HID, 0);   // semantic_q
    k_gemm<<<gT, 256, 0, stream>>>(ucX, Wu1, bu1, nullptr, uh, TOK, SEM, 1);
    k_gemm<<<gT, 256, 0, stream>>>(uh,  Wu2, uv_b2, uco, nullptr, TOK, HID, 0);   // unit_context

    // 4) attention + state tower + heads
    k_attn<<<TOK, 256, 0, stream>>>(tqo, sqo, keym, valm, wnrm, eco,
                                    out + OFF_AT, out + OFF_EN, out + OFF_MX);
    k_state<<<TOK, 256, 0, stream>>>(tqo, sqo, eco, uco, scal8, stX);
    k_gemm<<<gT, 256, 0, stream>>>(stX, Wst1, bst1, nullptr, sth, TOK, STP, 1);
    k_gemm<<<gT, 256, 0, stream>>>(sth, Wst2, st_b2, out + OFF_ACT, nullptr, TOK, HID, 1); // gelu -> activation_state
    k_heads<<<TOK / 8, 256, 0, stream>>>(out + OFF_ACT, h_al_w, h_al_b, h_ut_w, h_ut_b,
                                         h_be_w, h_be_b, out + OFF_AL, out + OFF_UT, out + OFF_BE);
}
